// Block_24747601560233
// MI455X (gfx1250) — matrix operand order repaired and checked by reading, not compiled, not run on hardware
//
#include <hip/hip_runtime.h>
#include <hip/hip_bf16.h>
#include <math.h>

typedef __attribute__((ext_vector_type(16))) _Float16 v16h;
typedef __attribute__((ext_vector_type(8)))  float    v8f;

#define DIMC    768
#define HEADS   12
#define HD      64
#define MLPH    3072
#define NTOK    197
#define BATCH   64
#define MROWS   (BATCH * NTOK)   // 12608 = 197 * 64
#define QKVN    (3 * DIMC)       // 2304
#define KTILES  13               // ceil(197/16) -> 208 padded keys
#define KPAD    224              // padded to multiple of 32 for AV K-loop

union FragU { uint4 u[2]; v16h h; };

// A operand (16x32, f16): lane holds row m=lane%16; g=lane/16 selects K groups
// {g*8..g*8+7} and {16+g*8..16+g*8+7}. Caller passes p = rowBase + k0 + g*8.
__device__ __forceinline__ v16h ldfragA(const _Float16* p) {
  FragU f;
  f.u[0] = *(const uint4*)(p);
  f.u[1] = *(const uint4*)(p + 16);
  return f.h;
}
// B operand (32x16, f16): lane holds col n=lane%16; g=lane/16 selects K groups
// {g*8..g*8+7} and {16+g*8..16+g*8+7}. Caller passes p = colRowBase + k0 + g*8.
__device__ __forceinline__ v16h ldfragB(const _Float16* p) {
  FragU f;
  f.u[0] = *(const uint4*)(p);
  f.u[1] = *(const uint4*)(p + 16);
  return f.h;
}

// ---------------------------------------------------------------- conversions
__global__ void cvt_f32_f16(const float* __restrict__ s, _Float16* __restrict__ d, int n) {
  int i = blockIdx.x * 256 + threadIdx.x;
  if (i < n) d[i] = (_Float16)s[i];
}

// --------------------------------------------------- rel-pos bias precompute
__global__ void bias_pre_kernel(const int* __restrict__ idx,
                                const float* __restrict__ tab,
                                float* __restrict__ bias) {
  int t = blockIdx.x * 256 + threadIdx.x;       // over 197*197
  if (t < NTOK * NTOK) {
    int id = idx[t];
    #pragma unroll
    for (int h = 0; h < HEADS; h++)
      bias[(size_t)h * NTOK * NTOK + t] = tab[id * HEADS + h];
  }
}

// ------------------------------------------------------------------ layernorm
__global__ __launch_bounds__(256) void ln_kernel(const float* __restrict__ x,
                                                 const float* __restrict__ w,
                                                 const float* __restrict__ b,
                                                 _Float16* __restrict__ out) {
  __shared__ float r1[256], r2[256];
  const int row = blockIdx.x;
  const float* xr = x + (size_t)row * DIMC;
  float s = 0.f, q = 0.f;
  for (int i = threadIdx.x; i < DIMC; i += 256) { float v = xr[i]; s += v; q += v * v; }
  r1[threadIdx.x] = s; r2[threadIdx.x] = q;
  __syncthreads();
  for (int st = 128; st > 0; st >>= 1) {
    if (threadIdx.x < st) {
      r1[threadIdx.x] += r1[threadIdx.x + st];
      r2[threadIdx.x] += r2[threadIdx.x + st];
    }
    __syncthreads();
  }
  float mean = r1[0] * (1.0f / DIMC);
  float var  = r2[0] * (1.0f / DIMC) - mean * mean;
  float rs   = rsqrtf(var + 1e-5f);
  for (int i = threadIdx.x; i < DIMC; i += 256)
    out[(size_t)row * DIMC + i] = (_Float16)((xr[i] - mean) * rs * w[i] + b[i]);
}

// --------------------------------------------------------------- WMMA GEMM
// Y = A(MxK,f16) @ W(NxK,f16)^T, 4 waves/block, wave tile 64x64, block 64x256.
// MODE 0: qkv  -> half out + concat(q_bias,0,v_bias)
// MODE 1: proj -> f32 x1 = resid + gamma*(acc + bias0)
// MODE 2: fc1  -> half out = gelu(acc + bias0)
// MODE 3: fc2  -> f = gamma*(acc+bias0); out_f0 = resid + f; out_f1 = f
template<int MODE>
__global__ __launch_bounds__(128) void gemm_wmma_kernel(
    const _Float16* __restrict__ A, const _Float16* __restrict__ W,
    int K, int Nld,
    const float* __restrict__ bias0, const float* __restrict__ bias1,
    const float* __restrict__ gamma, const float* __restrict__ resid,
    float* __restrict__ out_f0, float* __restrict__ out_f1,
    _Float16* __restrict__ out_h) {
  const int lane = threadIdx.x & 31;
  const int wv   = threadIdx.x >> 5;
  const int g    = lane >> 4, l16 = lane & 15;
  const int m0   = blockIdx.y * 64;
  const int n0   = blockIdx.x * 256 + wv * 64;

  v8f acc[4][4];
  #pragma unroll
  for (int i = 0; i < 4; i++)
    #pragma unroll
    for (int j = 0; j < 4; j++) acc[i][j] = {};

  for (int kk = 0; kk < K; kk += 32) {
    v16h a[4];
    #pragma unroll
    for (int i = 0; i < 4; i++)
      a[i] = ldfragA(A + (size_t)(m0 + i * 16 + l16) * K + kk + g * 8);
    #pragma unroll
    for (int j = 0; j < 4; j++) {
      v16h bfr = ldfragB(W + (size_t)(n0 + j * 16 + l16) * K + kk + g * 8);
      #pragma unroll
      for (int i = 0; i < 4; i++)
        acc[i][j] = __builtin_amdgcn_wmma_f32_16x16x32_f16(
            false, a[i], false, bfr, (short)0, acc[i][j], false, false);
    }
  }

  #pragma unroll
  for (int i = 0; i < 4; i++) {
    #pragma unroll
    for (int j = 0; j < 4; j++) {
      #pragma unroll
      for (int r = 0; r < 8; r++) {
        const int row = m0 + i * 16 + r + 8 * g;
        const int col = n0 + j * 16 + l16;
        float v = acc[i][j][r];
        if (MODE == 0) {
          float bb = (col < DIMC) ? bias0[col]
                   : (col < 2 * DIMC ? 0.0f : bias1[col - 2 * DIMC]);
          out_h[(size_t)row * Nld + col] = (_Float16)(v + bb);
        } else if (MODE == 1) {
          float t = resid[(size_t)row * DIMC + col] + gamma[col] * (v + bias0[col]);
          out_f0[(size_t)row * DIMC + col] = t;
        } else if (MODE == 2) {
          float t  = v + bias0[col];
          float gl = 0.5f * t * (1.0f + erff(t * 0.70710678118654752440f));
          out_h[(size_t)row * Nld + col] = (_Float16)gl;
        } else {
          float f  = gamma[col] * (v + bias0[col]);
          float xo = resid[(size_t)row * DIMC + col] + f;
          out_f0[(size_t)row * DIMC + col] = xo;
          out_f1[(size_t)row * DIMC + col] = f;
        }
      }
    }
  }
}

// ----------------------------------------------------------------- attention
// One block (4 waves) per (head, batch). Scores + softmax + AV all via WMMA.
__global__ __launch_bounds__(128) void attn_kernel(
    const _Float16* __restrict__ qkv,   // [MROWS, 2304] f16
    const float* __restrict__ bias,     // [12,197,197] f32
    _Float16* __restrict__ out) {       // [MROWS, 768] f16
  __shared__ _Float16 vt[64 * KPAD];          // V transposed [d][key], 28 KB
  __shared__ _Float16 probs[4][16 * KPAD];    // per-wave probs, 28 KB
  const int h = blockIdx.x, b = blockIdx.y;
  const int tid = threadIdx.x;
  const int lane = tid & 31, wv = tid >> 5;
  const int g = lane >> 4, l16 = lane & 15;
  const size_t rowbase = (size_t)b * NTOK;

  // stage V transposed (zero-pad keys 197..223)
  for (int i = tid; i < 64 * KPAD; i += 128) {
    int d = i / KPAD, t = i % KPAD;
    _Float16 val = (_Float16)0.0f;
    if (t < NTOK) val = qkv[(rowbase + t) * QKVN + 2 * DIMC + h * HD + d];
    vt[d * KPAD + t] = val;
  }
  __syncthreads();

  _Float16* myp = &probs[wv][0];

  for (int qt = wv; qt < KTILES; qt += 4) {
    const int q0 = qt * 16;
    // -- Q fragments (K = head_dim = 64 -> two K-steps)
    v16h qa[2];
    {
      int qr = q0 + l16; if (qr > NTOK - 1) qr = NTOK - 1;
      const _Float16* qp = qkv + (rowbase + qr) * QKVN + h * HD;
      qa[0] = ldfragA(qp + 0 + g * 8);
      qa[1] = ldfragA(qp + 32 + g * 8);
    }
    // -- scores: q . k^T  (13 key tiles x 2 K-steps = 26 WMMAs)
    v8f acc[KTILES];
    #pragma unroll
    for (int kt = 0; kt < KTILES; kt++) acc[kt] = {};
    #pragma unroll
    for (int kt = 0; kt < KTILES; kt++) {
      int kr = kt * 16 + l16; if (kr > NTOK - 1) kr = NTOK - 1;
      const _Float16* kp = qkv + (rowbase + kr) * QKVN + DIMC + h * HD;
      #pragma unroll
      for (int s = 0; s < 2; s++) {
        v16h kb = ldfragB(kp + s * 32 + g * 8);
        acc[kt] = __builtin_amdgcn_wmma_f32_16x16x32_f16(
            false, qa[s], false, kb, (short)0, acc[kt], false, false);
      }
    }
    // -- scale + relative position bias + key mask
    #pragma unroll
    for (int kt = 0; kt < KTILES; kt++) {
      int key = kt * 16 + l16;
      #pragma unroll
      for (int r = 0; r < 8; r++) {
        int qq = q0 + r + 8 * g; if (qq > NTOK - 1) qq = NTOK - 1;
        if (key < NTOK)
          acc[kt][r] = acc[kt][r] * 0.125f + bias[((size_t)h * NTOK + qq) * NTOK + key];
        else
          acc[kt][r] = -INFINITY;
      }
    }
    // -- softmax in registers (rows live in 16-lane groups of each half-wave)
    float inv[8];
    #pragma unroll
    for (int r = 0; r < 8; r++) {
      float m = -INFINITY;
      #pragma unroll
      for (int kt = 0; kt < KTILES; kt++) m = fmaxf(m, acc[kt][r]);
      m = fmaxf(m, __shfl_xor(m, 1));
      m = fmaxf(m, __shfl_xor(m, 2));
      m = fmaxf(m, __shfl_xor(m, 4));
      m = fmaxf(m, __shfl_xor(m, 8));
      float s = 0.f;
      #pragma unroll
      for (int kt = 0; kt < KTILES; kt++) {
        float e = __expf(acc[kt][r] - m);
        acc[kt][r] = e; s += e;
      }
      s += __shfl_xor(s, 1);
      s += __shfl_xor(s, 2);
      s += __shfl_xor(s, 4);
      s += __shfl_xor(s, 8);
      inv[r] = 1.0f / s;
    }
    // -- C-layout -> LDS (row-major probs, zero-pad cols 208..223)
    #pragma unroll
    for (int r = 0; r < 8; r++) {
      int m = r + 8 * g;
      #pragma unroll
      for (int kt = 0; kt < KTILES; kt++)
        myp[m * KPAD + kt * 16 + l16] = (_Float16)(acc[kt][r] * inv[r]);
      myp[m * KPAD + 208 + l16] = (_Float16)0.0f;
    }
    // -- AV: probs(16x224) . V(224x64)  (7 K-steps x 4 d-tiles = 28 WMMAs)
    v8f oacc[4];
    #pragma unroll
    for (int dt = 0; dt < 4; dt++) oacc[dt] = {};
    #pragma unroll
    for (int s = 0; s < 7; s++) {
      v16h pa = ldfragA(myp + l16 * KPAD + s * 32 + g * 8);
      #pragma unroll
      for (int dt = 0; dt < 4; dt++) {
        v16h vb = ldfragB(vt + (dt * 16 + l16) * KPAD + s * 32 + g * 8);
        oacc[dt] = __builtin_amdgcn_wmma_f32_16x16x32_f16(
            false, pa, false, vb, (short)0, oacc[dt], false, false);
      }
    }
    // -- write attn output (heads concatenated), f16 for proj GEMM
    #pragma unroll
    for (int dt = 0; dt < 4; dt++) {
      #pragma unroll
      for (int r = 0; r < 8; r++) {
        int qq = q0 + r + 8 * g;
        if (qq < NTOK)
          out[(rowbase + qq) * DIMC + h * HD + dt * 16 + l16] = (_Float16)oacc[dt][r];
      }
    }
  }
}

// ------------------------------------------------------------------- launch
extern "C" void kernel_launch(void* const* d_in, const int* in_sizes, int n_in,
                              void* d_out, int out_size, void* d_ws, size_t ws_size,
                              hipStream_t stream) {
  const float* x         = (const float*)d_in[0];
  const int*   rel_idx   = (const int*)  d_in[1];
  const float* qkv_w     = (const float*)d_in[2];
  const float* q_bias    = (const float*)d_in[3];
  const float* v_bias    = (const float*)d_in[4];
  const float* rpb_table = (const float*)d_in[5];
  const float* proj_w    = (const float*)d_in[6];
  const float* proj_b    = (const float*)d_in[7];
  const float* n1_w      = (const float*)d_in[8];
  const float* n1_b      = (const float*)d_in[9];
  const float* n2_w      = (const float*)d_in[10];
  const float* n2_b      = (const float*)d_in[11];
  const float* fc1_w     = (const float*)d_in[12];
  const float* fc1_b     = (const float*)d_in[13];
  const float* fc2_w     = (const float*)d_in[14];
  const float* fc2_b     = (const float*)d_in[15];
  const float* gamma1    = (const float*)d_in[16];
  const float* gamma2    = (const float*)d_in[17];
  float* out_x = (float*)d_out;
  float* out_f = (float*)d_out + (size_t)MROWS * DIMC;

  // ---- workspace carve (≈248 MB)
  char* p = (char*)d_ws;
  auto alloc = [&](size_t bytes) -> void* {
    void* r = (void*)p; p += (bytes + 255) & ~(size_t)255; return r;
  };
  _Float16* wqkv    = (_Float16*)alloc((size_t)QKVN * DIMC * 2);
  _Float16* wproj   = (_Float16*)alloc((size_t)DIMC * DIMC * 2);
  _Float16* wfc1    = (_Float16*)alloc((size_t)MLPH * DIMC * 2);
  _Float16* wfc2    = (_Float16*)alloc((size_t)DIMC * MLPH * 2);
  _Float16* hbuf    = (_Float16*)alloc((size_t)MROWS * DIMC * 2);
  _Float16* qkvbuf  = (_Float16*)alloc((size_t)MROWS * QKVN * 2);
  float*    biasp   = (float*)   alloc((size_t)HEADS * NTOK * NTOK * 4);
  _Float16* attnout = (_Float16*)alloc((size_t)MROWS * DIMC * 2);
  float*    x1      = (float*)   alloc((size_t)MROWS * DIMC * 4);
  _Float16* h2buf   = (_Float16*)alloc((size_t)MROWS * DIMC * 2);
  _Float16* hid     = (_Float16*)alloc((size_t)MROWS * MLPH * 2);

  // ---- weight conversion fp32 -> f16
  auto cvt = [&](const float* s, _Float16* d, int n) {
    cvt_f32_f16<<<(n + 255) / 256, 256, 0, stream>>>(s, d, n);
  };
  cvt(qkv_w, wqkv, QKVN * DIMC);
  cvt(proj_w, wproj, DIMC * DIMC);
  cvt(fc1_w, wfc1, MLPH * DIMC);
  cvt(fc2_w, wfc2, DIMC * MLPH);

  // ---- rel-pos bias table [12,197,197]
  bias_pre_kernel<<<(NTOK * NTOK + 255) / 256, 256, 0, stream>>>(rel_idx, rpb_table, biasp);

  // ---- LN1 -> h (f16)
  ln_kernel<<<MROWS, 256, 0, stream>>>(x, n1_w, n1_b, hbuf);

  // ---- qkv = h @ qkv_w^T + concat(q_bias,0,v_bias)   [12608 x 2304]
  gemm_wmma_kernel<0><<<dim3(QKVN / 256, MROWS / 64), 128, 0, stream>>>(
      hbuf, wqkv, DIMC, QKVN, q_bias, v_bias, nullptr, nullptr,
      nullptr, nullptr, qkvbuf);

  // ---- attention per (head, batch)
  attn_kernel<<<dim3(HEADS, BATCH), 128, 0, stream>>>(qkvbuf, biasp, attnout);

  // ---- x1 = x + gamma1 * (attn_out @ proj_w^T + proj_b)
  gemm_wmma_kernel<1><<<dim3(DIMC / 256, MROWS / 64), 128, 0, stream>>>(
      attnout, wproj, DIMC, DIMC, proj_b, nullptr, gamma1, x,
      x1, nullptr, nullptr);

  // ---- LN2 -> h2 (f16)
  ln_kernel<<<MROWS, 256, 0, stream>>>(x1, n2_w, n2_b, h2buf);

  // ---- hid = gelu(h2 @ fc1_w^T + fc1_b)   [12608 x 3072]
  gemm_wmma_kernel<2><<<dim3(MLPH / 256, MROWS / 64), 128, 0, stream>>>(
      h2buf, wfc1, DIMC, MLPH, fc1_b, nullptr, nullptr, nullptr,
      nullptr, nullptr, hid);

  // ---- f = gamma2*(hid @ fc2_w^T + fc2_b); out_x = x1 + f; out_f = f
  gemm_wmma_kernel<3><<<dim3(DIMC / 256, MROWS / 64), 128, 0, stream>>>(
      hid, wfc2, MLPH, DIMC, fc2_b, nullptr, gamma2, x1,
      out_x, out_f, nullptr);

  (void)in_sizes; (void)n_in; (void)out_size; (void)ws_size;
}
